// Det_AllHist_GLM_17806934409620
// MI455X (gfx1250) — compile-verified, matrix operand order checked by reading, not run on hardware
//
#include <hip/hip_runtime.h>
#include <hip/hip_bf16.h>

typedef __attribute__((ext_vector_type(16))) _Float16     v16h;
typedef __attribute__((ext_vector_type(8)))  float        v8f;
typedef __attribute__((ext_vector_type(4)))  unsigned int v4u;
typedef __attribute__((ext_vector_type(8)))  int          v8i;
typedef __attribute__((ext_vector_type(4)))  int          v4i;

#define T_DATA   20000
#define T_NO     200
#define SUB_NO   20
#define E_NO     2000
#define I_NO     500
#define E_PAD    2016   // 63 * 32
#define I_PAD    512    // 16 * 32
#define NPAD     32

#if defined(__has_builtin)
#if __has_builtin(__builtin_amdgcn_tensor_load_to_lds)
#define HAS_TDM 1
#endif
#endif
#ifndef HAS_TDM
#define HAS_TDM 0
#endif

__device__ __forceinline__ float sigm(float x) { return 1.f / (1.f + expf(-x)); }

#if HAS_TDM
// ---------------------------------------------------------------------------
// Issue a TDM 2D tile load (f32 elements) global -> LDS.
// D# per CDNA5 ISA:
//   group0: [1:0]=count=1 | lds_addr[63:32] | global_addr[120:64] | type=2
//   group1: data_size=2 (4B) @ bits17:16, tensor_dim0@[79:48],
//           tensor_dim1@[111:80], tile_dim0@[127:112], tile_dim1@[143:128],
//           tensor_dim0_stride@[207:160]
// OOB columns (x >= dim0) read as zero -> replaces per-element K guards.
// ---------------------------------------------------------------------------
__device__ __forceinline__ void tdm_load_2d_f32(unsigned lds_off, const float* gptr,
                                                unsigned dim0, unsigned dim1,
                                                unsigned stride0) {
    unsigned long long ga = (unsigned long long)(uintptr_t)gptr;
    v4u g0;
    g0.x = 1u;                                            // count=1 (valid)
    g0.y = lds_off;                                       // lds_addr (bytes)
    g0.z = (unsigned)(ga & 0xFFFFFFFFu);                  // global_addr[31:0]
    g0.w = (unsigned)((ga >> 32) & 0x01FFFFFFu) | 0x80000000u; // [56:32]|type=2
    v8i g1;
    g1[0] = (int)(2u << 16);                              // data_size=4B, mask=0
    g1[1] = (int)((dim0 & 0xFFFFu) << 16);                // tensor_dim0[15:0]
    g1[2] = (int)(((dim0 >> 16) & 0xFFFFu) | ((dim1 & 0xFFFFu) << 16));
    g1[3] = (int)(((dim1 >> 16) & 0xFFFFu) | (32u << 16));// tile_dim0 = 32
    g1[4] = (int)16u;                                     // tile_dim1=16, tile_dim2=0
    g1[5] = (int)stride0;                                 // dim0_stride[31:0]
    g1[6] = 0;                                            // stride hi / dim1_stride
    g1[7] = 0;
    v4i z4 = {0, 0, 0, 0};
#if __clang_major__ >= 23
    v8i z8 = {0, 0, 0, 0, 0, 0, 0, 0};
    __builtin_amdgcn_tensor_load_to_lds(g0, g1, z4, z4, z8, 0);
#else
    __builtin_amdgcn_tensor_load_to_lds(g0, g1, z4, z4, 0);
#endif
}
#endif

// ---------------------------------------------------------------------------
// Build padded transposed f16 B matrix:  Bt[n][k] = C[n*K + k]  (zero padded)
// ---------------------------------------------------------------------------
__global__ void build_B_kernel(const float* __restrict__ C, int K, int Kpad,
                               _Float16* __restrict__ Bt) {
    int idx = blockIdx.x * 256 + threadIdx.x;
    if (idx >= NPAD * Kpad) return;
    int n = idx / Kpad, k = idx % Kpad;
    float v = (n < SUB_NO && k < K) ? C[n * K + k] : 0.f;
    Bt[idx] = (_Float16)v;
}

// ---------------------------------------------------------------------------
// Build all temporal filters and the out_filters output block (80 x 200).
// ---------------------------------------------------------------------------
__global__ void build_filters_kernel(
    const float* __restrict__ Tau_syn, const float* __restrict__ Delta_syn,
    const float* __restrict__ W_syn,  const float* __restrict__ Tau_spk,
    const float* __restrict__ W_spk,  const float* __restrict__ W_hist,
    const float* __restrict__ Tau_out, const float* __restrict__ W_out,
    float* __restrict__ out_filt,
    float* __restrict__ ek, float* __restrict__ ik,
    float* __restrict__ sk, float* __restrict__ hk,
    float* __restrict__ ok) {
    const float PI = 3.14159265358979323846f;
    for (int idx = threadIdx.x; idx < 80 * T_NO; idx += blockDim.x) {
        int row = idx / T_NO, col = idx % T_NO;
        float t = (float)col;
        float v;
        if (row < 20) {
            int s = row;
            float te  = fmaxf(t - expf(Delta_syn[s * 2 + 0]), 0.f);
            float tte = te * sigm(Tau_syn[s * 2 + 0]);
            v = tte * expf(-tte) * expf(W_syn[s * 2 + 0]);
            ek[s * T_NO + col] = v;
        } else if (row < 40) {
            int s = row - 20;
            float ti  = fmaxf(t - expf(Delta_syn[s * 2 + 1]), 0.f);
            float tti = ti * sigm(Tau_syn[s * 2 + 1]);
            v = -tti * expf(-tti) * expf(W_syn[s * 2 + 1]);
            ik[s * T_NO + col] = v;
        } else if (row < 60) {
            int s = row - 40;
            float tts = t * sigm(Tau_spk[s]);
            v = tts * expf(-tts) * expf(W_spk[s]);
            sk[s * T_NO + col] = v;
        } else {
            int s = row - 60;
            float raw = 4.f * logf(t + 1.f);
            float acc = 0.f;
            #pragma unroll
            for (int i = 0; i < 17; ++i) {
                float phi = 0.5f * PI * (float)i;
                float d = raw - phi;
                float b = (d < -PI || d > PI) ? 0.f : (0.5f * cosf(d) + 0.5f);
                acc += W_hist[s * 17 + i] * b;
            }
            v = acc;
            hk[s * T_NO + col] = v;
        }
        out_filt[idx] = v;
    }
    if (threadIdx.x < T_NO) {
        float t  = (float)threadIdx.x;
        float to = t * sigm(Tau_out[0]);
        ok[threadIdx.x] = to * expf(-to) * expf(W_out[0]);
    }
}

// ---------------------------------------------------------------------------
// WMMA GEMM:  out[t][n] = sum_k S[t][k] * Bt[n][k]
// One wave per block -> 16 x 32 output tile.  A tile (16x32 f32) is staged in
// LDS via the Tensor Data Mover (double buffered, s_wait_tensorcnt), consumed
// as unconditional ds_load_b128 + cvt to f16.  B read directly (b128 loads).
// Layouts per CDNA5 ISA §7.12.2 (wave32).
// ---------------------------------------------------------------------------
__global__ void gemm_wmma_kernel(const float* __restrict__ S, int K, int Kpad,
                                 const _Float16* __restrict__ Bt,
                                 float* __restrict__ out /* [T][32] */) {
    __shared__ float Atile[2][16 * 32];

    int lane = threadIdx.x;                 // 0..31
    int m    = lane & 15;
    int hi   = lane >> 4;
    int koff = hi * 8;
    int t0   = blockIdx.x * 16;
    const _Float16* b0r = Bt + (size_t)m * Kpad;
    const _Float16* b1r = Bt + (size_t)(m + 16) * Kpad;
    const float* abase  = S + (size_t)t0 * K;

    v8f c0 = {}, c1 = {};
    const int nk = Kpad / 32;

#if HAS_TDM
    unsigned lds0 = (unsigned)(uintptr_t)&Atile[0][0];
    unsigned lds1 = (unsigned)(uintptr_t)&Atile[1][0];
    tdm_load_2d_f32(lds0, abase, (unsigned)K, (unsigned)(T_DATA - t0), (unsigned)K);
#endif

    for (int i = 0; i < nk; ++i) {
        int kt = i * 32;
#if HAS_TDM
        if (i + 1 < nk) {
            unsigned rem = (unsigned)(K - (kt + 32) > 0 ? K - (kt + 32) : 0);
            tdm_load_2d_f32((i & 1) ? lds0 : lds1, abase + kt + 32,
                            rem, (unsigned)(T_DATA - t0), (unsigned)K);
            __builtin_amdgcn_s_wait_tensorcnt(1);   // tile i complete
        } else {
            __builtin_amdgcn_s_wait_tensorcnt(0);
        }
#else
        // branchless cooperative stage: clamp index, select after load
        __syncthreads();
        #pragma unroll
        for (int e = lane; e < 16 * 32; e += 32) {
            int r = e >> 5, c = e & 31;
            int kk = kt + c;
            int kc = kk < K - 1 ? kk : K - 1;
            float v = abase[(size_t)r * K + kc];
            Atile[0][e] = (kk < K) ? v : 0.f;
        }
        __syncthreads();
#endif
        const float* Af =
#if HAS_TDM
            &Atile[i & 1][0];
#else
            &Atile[0][0];
#endif
        if (kt + 64 < Kpad) __builtin_prefetch(b0r + kt + 64, 0, 0);

        v16h a;
        const float* arow = Af + m * 32 + koff;
        #pragma unroll
        for (int j = 0; j < 8; ++j) {
            a[j]     = (_Float16)arow[j];
            a[j + 8] = (_Float16)arow[16 + j];
        }
        int kb = kt + koff;
        v16h b0, b1;
        #pragma unroll
        for (int j = 0; j < 8; ++j) {
            b0[j]     = b0r[kb + j];
            b0[j + 8] = b0r[kb + 16 + j];
            b1[j]     = b1r[kb + j];
            b1[j + 8] = b1r[kb + 16 + j];
        }
        c0 = __builtin_amdgcn_wmma_f32_16x16x32_f16(false, a, false, b0,
                                                    (short)0, c0, false, false);
        c1 = __builtin_amdgcn_wmma_f32_16x16x32_f16(false, a, false, b1,
                                                    (short)0, c1, false, false);
    }

    int n  = lane & 15;
    int mb = hi * 8;
    #pragma unroll
    for (int r = 0; r < 8; ++r) {
        int trow = t0 + mb + r;
        out[(size_t)trow * NPAD + n]      = c0[r];
        out[(size_t)trow * NPAD + 16 + n] = c1[r];
    }
}

// ---------------------------------------------------------------------------
// Causal per-subunit 200-tap convolution.
// ---------------------------------------------------------------------------
__global__ void conv_syn_kernel(const float* __restrict__ synE,
                                const float* __restrict__ synI,
                                const float* __restrict__ ek,
                                const float* __restrict__ ik,
                                float* __restrict__ syn) {
    int s = threadIdx.x;
    int t = blockIdx.x * 8 + threadIdx.y;
    if (t >= T_DATA) return;
    float acc = 0.f;
    if (s < SUB_NO) {
        int lmax = (t < T_NO) ? t : T_NO;
        const float* eks = ek + s * T_NO;
        const float* iks = ik + s * T_NO;
        for (int l = 0; l < lmax; ++l) {
            int idx = t - 1 - l;
            acc += synE[(size_t)idx * NPAD + s] * eks[l]
                 + synI[(size_t)idx * NPAD + s] * iks[l];
        }
    }
    syn[(size_t)t * NPAD + s] = acc;
}

// ---------------------------------------------------------------------------
// Sequential scan: one wave32 block, LDS ring + LDS-cached kernels.
// ---------------------------------------------------------------------------
__global__ void scan_kernel(const float* __restrict__ syn,
                            const float* __restrict__ C_den,
                            const float* __restrict__ Theta,
                            const float* __restrict__ spkk,
                            const float* __restrict__ histk,
                            float* __restrict__ outZ,
                            float* __restrict__ outP) {
    __shared__ float Zring[T_NO * 32];
    __shared__ float spk_kc[SUB_NO * T_NO];
    __shared__ float hist_kc[SUB_NO * T_NO];
    __shared__ float Cd[SUB_NO * SUB_NO];
    __shared__ float spk_sum[32];

    int lane = threadIdx.x;
    for (int i = lane; i < T_NO * 32; i += 32) Zring[i] = 0.f;
    for (int i = lane; i < SUB_NO * T_NO; i += 32) {
        spk_kc[i]  = spkk[i];
        hist_kc[i] = histk[i];
    }
    for (int i = lane; i < SUB_NO * SUB_NO; i += 32) Cd[i] = C_den[i];
    __syncthreads();

    float th = (lane < SUB_NO) ? Theta[lane] : 0.f;
    int head = 0;
    for (int t = 0; t < T_DATA; ++t) {
        float h = 0.f, q = 0.f;
        if (lane < SUB_NO) {
            const float* hks = hist_kc + lane * T_NO;
            const float* sks = spk_kc  + lane * T_NO;
            int p = head - 1; if (p < 0) p += T_NO;
            for (int a = 0; a < T_NO; ++a) {
                float z = Zring[p * 32 + lane];
                h += z * hks[a];
                q += z * sks[a];
                --p; if (p < 0) p = T_NO - 1;
            }
        }
        spk_sum[lane] = q;
        __syncthreads();

        float spike = 0.f;
        if (lane < SUB_NO) {
            float si = 0.f;
            #pragma unroll 4
            for (int j = 0; j < SUB_NO; ++j) si += Cd[lane * SUB_NO + j] * spk_sum[j];
            float sub_in = si + syn[(size_t)t * NPAD + lane] + th
                         + ((lane == 0) ? 0.f : h);
            spike = (sub_in > 0.f) ? 1.f : 0.f;
            outZ[(size_t)t * SUB_NO + lane] = spike;
            outP[(size_t)t * SUB_NO + lane] = 1.f / (1.f + expf(-sub_in));
        }
        __syncthreads();
        Zring[head * 32 + lane] = spike;
        ++head; if (head == T_NO) head = 0;
        __syncthreads();
    }
}

// ---------------------------------------------------------------------------
// V[t] = sum_l Z[t-1-l][0] * out_kern[l]
// ---------------------------------------------------------------------------
__global__ void out_conv_kernel(const float* __restrict__ Z,
                                const float* __restrict__ ok,
                                float* __restrict__ V) {
    int t = blockIdx.x * 256 + threadIdx.x;
    if (t >= T_DATA) return;
    float acc = 0.f;
    int lmax = (t < T_NO) ? t : T_NO;
    for (int l = 0; l < lmax; ++l)
        acc += Z[(size_t)(t - 1 - l) * SUB_NO] * ok[l];
    V[t] = acc;
}

// ---------------------------------------------------------------------------
extern "C" void kernel_launch(void* const* d_in, const int* in_sizes, int n_in,
                              void* d_out, int out_size, void* d_ws, size_t ws_size,
                              hipStream_t stream) {
    const float* S_e       = (const float*)d_in[0];
    const float* S_i       = (const float*)d_in[1];
    const float* C_den     = (const float*)d_in[2];
    const float* C_syn_e   = (const float*)d_in[3];
    const float* C_syn_i   = (const float*)d_in[4];
    const float* Tau_syn   = (const float*)d_in[5];
    const float* Delta_syn = (const float*)d_in[6];
    const float* W_syn     = (const float*)d_in[7];
    const float* Tau_spk   = (const float*)d_in[8];
    const float* W_spk     = (const float*)d_in[9];
    const float* W_hist    = (const float*)d_in[10];
    const float* Theta     = (const float*)d_in[11];
    const float* Tau_out   = (const float*)d_in[12];
    const float* W_out     = (const float*)d_in[13];

    float* out  = (float*)d_out;
    float* V    = out;                 // 20000
    float* Z    = out + 20000;         // 400000
    float* P    = out + 420000;        // 400000
    float* filt = out + 820000;        // 16000

    char* ws = (char*)d_ws;
    _Float16* Bt_e = (_Float16*)(ws + 0);          // 32*2016*2 = 129024 B
    _Float16* Bt_i = (_Float16*)(ws + 131072);     // 32*512*2  =  32768 B
    float* synE = (float*)(ws + 163840);           // 20000*32*4
    float* synI = (float*)(ws + 163840 + 2560000);
    float* syn  = (float*)(ws + 163840 + 2 * 2560000);
    float* ek   = (float*)(ws + 163840 + 3 * 2560000);
    float* ik   = (float*)(ws + 163840 + 3 * 2560000 + 16000);
    float* sk   = (float*)(ws + 163840 + 3 * 2560000 + 32000);
    float* hk   = (float*)(ws + 163840 + 3 * 2560000 + 48000);
    float* ok   = (float*)(ws + 163840 + 3 * 2560000 + 64000);

    build_B_kernel<<<(NPAD * E_PAD + 255) / 256, 256, 0, stream>>>(C_syn_e, E_NO, E_PAD, Bt_e);
    build_B_kernel<<<(NPAD * I_PAD + 255) / 256, 256, 0, stream>>>(C_syn_i, I_NO, I_PAD, Bt_i);
    build_filters_kernel<<<1, 256, 0, stream>>>(Tau_syn, Delta_syn, W_syn, Tau_spk,
                                                W_spk, W_hist, Tau_out, W_out,
                                                filt, ek, ik, sk, hk, ok);
    gemm_wmma_kernel<<<T_DATA / 16, 32, 0, stream>>>(S_e, E_NO, E_PAD, Bt_e, synE);
    gemm_wmma_kernel<<<T_DATA / 16, 32, 0, stream>>>(S_i, I_NO, I_PAD, Bt_i, synI);
    conv_syn_kernel<<<T_DATA / 8, dim3(32, 8), 0, stream>>>(synE, synI, ek, ik, syn);
    scan_kernel<<<1, 32, 0, stream>>>(syn, C_den, Theta, sk, hk, Z, P);
    out_conv_kernel<<<(T_DATA + 255) / 256, 256, 0, stream>>>(Z, ok, V);
}
